// PatchSVAE_83004537962967
// MI455X (gfx1250) — compile-verified, matrix operand order checked by reading, not run on hardware
//
#include <hip/hip_runtime.h>

// ================= types / helpers =================
typedef __attribute__((ext_vector_type(16))) __bf16 v16bf;
typedef __attribute__((ext_vector_type(8)))  float  v8f;
typedef __attribute__((ext_vector_type(2)))  float  v2f;

union BFrag { v16bf v; unsigned int u[8]; uint4 q[2]; };

__device__ __forceinline__ unsigned short f2bf(float f) {
  unsigned int u = __float_as_uint(f);
  unsigned int r = (u + 0x7fffu + ((u >> 16) & 1u)) >> 16;
  return (unsigned short)r;
}
__device__ __forceinline__ float gelu_f(float x) {
  return 0.5f * x * (1.0f + erff(x * 0.70710678118654752440f));
}

#ifndef __has_builtin
#define __has_builtin(x) 0
#endif
#if __has_builtin(__builtin_amdgcn_wmma_f32_16x16x4_f32)
#define HAVE_WMMA_F32 1
#else
#define HAVE_WMMA_F32 0
#endif

// ---- CDNA5 async global->LDS copy (16B per lane, ASYNCcnt-tracked) ----
__device__ __forceinline__ void async_copy_b128(unsigned int lds_off, const void* gaddr) {
  asm volatile("global_load_async_to_lds_b128 %0, %1, off"
               :
               : "v"(lds_off), "v"((unsigned long long)(size_t)gaddr)
               : "memory");
}
__device__ __forceinline__ void wait_async0() {
#if __has_builtin(__builtin_amdgcn_s_wait_asynccnt)
  __builtin_amdgcn_s_wait_asynccnt(0);
#else
  asm volatile("s_wait_asynccnt 0x0" ::: "memory");
#endif
}
__device__ __forceinline__ void wait_async1() {
#if __has_builtin(__builtin_amdgcn_s_wait_asynccnt)
  __builtin_amdgcn_s_wait_asynccnt(1);
#else
  asm volatile("s_wait_asynccnt 0x1" ::: "memory");
#endif
}

// ================= f32 -> bf16 convert =================
__global__ void cvt_f32_bf16_k(const float* __restrict__ in, unsigned short* __restrict__ out, int n) {
  int i = blockIdx.x * blockDim.x + threadIdx.x;
  if (i < n) out[i] = f2bf(in[i]);
}

// ================= patchify: x(16,3,256,256) -> pt_bf16 (4096,768) =================
__global__ void patchify_k(const float* __restrict__ x, unsigned short* __restrict__ pt) {
  int idx = blockIdx.x * blockDim.x + threadIdx.x;
  if (idx >= 4096 * 768) return;
  int R = idx / 768, col = idx % 768;
  int b = R / 256, n = R % 256;
  int gy = n / 16, gx = n % 16;
  int c = col / 256, r = col % 256;
  int py = r / 16, px = r % 16;
  float v = x[(((size_t)(b * 3 + c) * 256 + (gy * 16 + py)) * 256) + (gx * 16 + px)];
  pt[idx] = f2bf(v);
}

// ================= stitch: pat(4096,768) -> img(16,3,256,256) =================
__global__ void stitch_k(const float* __restrict__ pat, float* __restrict__ img) {
  int idx = blockIdx.x * blockDim.x + threadIdx.x;
  if (idx >= 16 * 3 * 256 * 256) return;
  int w = idx & 255, h = (idx >> 8) & 255;
  int c = (idx >> 16) % 3, b = idx / (3 * 65536);
  int n = (h / 16) * 16 + (w / 16);
  int col = c * 256 + (h % 16) * 16 + (w % 16);
  img[idx] = pat[((size_t)(b * 256 + n)) * 768 + col];
}

// ================= LayerNorm (rows of D) -> bf16 =================
__global__ void layernorm_k(const float* __restrict__ x, const float* __restrict__ g,
                            const float* __restrict__ b, unsigned short* __restrict__ out, int D) {
  int row = blockIdx.x;
  const float* xr = x + (size_t)row * D;
  float s = 0.f, s2 = 0.f;
  for (int i = threadIdx.x; i < D; i += blockDim.x) { float v = xr[i]; s += v; s2 += v * v; }
  for (int off = 16; off > 0; off >>= 1) { s += __shfl_down(s, off); s2 += __shfl_down(s2, off); }
  __shared__ float rs[8], rs2[8];
  int wid = threadIdx.x >> 5;
  if ((threadIdx.x & 31) == 0) { rs[wid] = s; rs2[wid] = s2; }
  __syncthreads();
  if (threadIdx.x == 0) {
    float S = 0.f, S2 = 0.f;
    for (int i = 0; i < (int)(blockDim.x >> 5); ++i) { S += rs[i]; S2 += rs2[i]; }
    rs[0] = S; rs2[0] = S2;
  }
  __syncthreads();
  float mean = rs[0] / D;
  float var = rs2[0] / D - mean * mean;
  float inv = rsqrtf(var + 1e-5f);
  for (int i = threadIdx.x; i < D; i += blockDim.x)
    out[(size_t)row * D + i] = f2bf((xr[i] - mean) * inv * g[i] + b[i]);
}

// ================= bf16 WMMA GEMM: Y = X(MxK) * W(N,K)^T + bias =================
// Block tile 128(M) x 64(N): 8 waves x (16x64). W tile double-buffered in LDS via
// global_load_async_to_lds_b128; fragments are 128-bit loads (2 per frag).
template <bool GELU, bool RES, bool OUT32, bool OUTBF>
__global__ __launch_bounds__(256) void gemm_bf16_k(
    const unsigned short* __restrict__ X, const unsigned short* __restrict__ W,
    const float* __restrict__ bias, const float* __restrict__ resid,
    float* __restrict__ outf, unsigned short* __restrict__ outb, int M, int N, int K) {
  // 64 rows x 32 K of bf16 per buffer; row padded to 40 ushorts (80B) so the 16
  // rows hit distinct LDS banks on ds_load_b128 (20 dwords/row, gcd(20,64) span).
  __shared__ __attribute__((aligned(16))) unsigned short Ws[2][64][40];
  int mblocks = M >> 7;
  int mb = blockIdx.x % mblocks, nb = blockIdx.x / mblocks;
  int wid = threadIdx.x >> 5, lane = threadIdx.x & 31;
  int m0 = (mb << 7) + (wid << 4);
  int n0 = nb << 6;
  if (n0 >= N) return;
  int mloc = lane & 15, half = lane >> 4;
  int mrow = m0 + mloc;

  // staging assignment: thread t stages 16B: W row n0+(t>>2), K offset (t&3)*8
  int sr = threadIdx.x >> 2;
  int sk = (threadIdx.x & 3) * 8;
  const unsigned short* gW = W + (size_t)(n0 + sr) * K + sk;
  unsigned int lds_off[2];
  lds_off[0] = (unsigned int)(size_t)&Ws[0][sr][sk];
  lds_off[1] = (unsigned int)(size_t)&Ws[1][sr][sk];

  const unsigned short* xrow = X + (size_t)mrow * K;
  int nIter = K >> 5;
  v8f acc[4] = {};

  async_copy_b128(lds_off[0], gW);  // stage tile 0
  for (int it = 0; it < nIter; ++it) {
    int k0 = it << 5;
    if (it + 1 < nIter) {
      async_copy_b128(lds_off[(it + 1) & 1], gW + k0 + 32);  // stream next tile
      wait_async1();                                          // tile `it` landed
    } else {
      wait_async0();
    }
    __syncthreads();
    BFrag a;
    a.q[0] = *(const uint4*)(xrow + k0 + 8 * half);
    a.q[1] = *(const uint4*)(xrow + k0 + 16 + 8 * half);
    const unsigned short* base = &Ws[it & 1][0][0];
#pragma unroll
    for (int j = 0; j < 4; ++j) {
      BFrag bf;
      const unsigned short* wr = base + (j * 16 + mloc) * 40;
      bf.q[0] = *(const uint4*)(wr + 8 * half);
      bf.q[1] = *(const uint4*)(wr + 16 + 8 * half);
      acc[j] = __builtin_amdgcn_wmma_f32_16x16x32_bf16(false, a.v, false, bf.v, (short)0, acc[j], false, false);
    }
    __syncthreads();
  }
#pragma unroll
  for (int j = 0; j < 4; ++j) {
    int n = n0 + j * 16 + mloc;
    float bv = bias[n];
#pragma unroll
    for (int i = 0; i < 8; ++i) {
      int m = m0 + i + 8 * half;
      float val = acc[j][i] + bv;
      if (GELU) val = gelu_f(val);
      if (RES) val += resid[(size_t)m * N + n];
      if (OUT32) outf[(size_t)m * N + n] = val;
      if (OUTBF) outb[(size_t)m * N + n] = f2bf(val);
    }
  }
}

// ================= Gram: G = A^T A per matrix via f32 WMMA (A: 256x16) =================
__global__ __launch_bounds__(256) void gram_k(const float* __restrict__ A, float* __restrict__ G) {
  int wave = (blockIdx.x * blockDim.x + threadIdx.x) >> 5;  // = matrix id
  int lane = threadIdx.x & 31;
  const float* Ab = A + (size_t)wave * 4096;
  int d = lane & 15, half = lane >> 4;
#if HAVE_WMMA_F32
  v8f acc = {};
  for (int k0 = 0; k0 < 256; k0 += 4) {
    v2f a;
    a.x = Ab[(k0 + 2 * half) * 16 + d];
    a.y = Ab[(k0 + 2 * half + 1) * 16 + d];
    acc = __builtin_amdgcn_wmma_f32_16x16x4_f32(false, a, false, a, (short)0, acc, false, false);
  }
#pragma unroll
  for (int i = 0; i < 8; ++i) G[(size_t)wave * 256 + (i + 8 * half) * 16 + d] = acc[i];
#else
  for (int i = lane; i < 256; i += 32) {
    int r = i >> 4, c = i & 15;
    float s = 0.f;
    for (int k = 0; k < 256; ++k) s += Ab[k * 16 + r] * Ab[k * 16 + c];
    G[(size_t)wave * 256 + i] = s;
  }
#endif
}

// ================= fp64 Jacobi eigh of 16x16 symmetric G, one wave/matrix =================
__global__ void jacobi_eigh_k(const float* __restrict__ Gin, float* __restrict__ Vout,
                              float* __restrict__ Sout) {
  int mat = blockIdx.x;
  int l = threadIdx.x;  // 32 lanes
  __shared__ double G[16][17];
  __shared__ double Vv[16][17];
  __shared__ double cs[2];
  __shared__ double evals[16];
  __shared__ int perm[16];
  for (int i = l; i < 256; i += 32) {
    int r = i >> 4, c = i & 15;
    double g = (double)Gin[(size_t)mat * 256 + i];
    if (r == c) g += 1e-12;
    G[r][c] = g;
    Vv[r][c] = (r == c) ? 1.0 : 0.0;
  }
  __syncthreads();
  for (int sweep = 0; sweep < 10; ++sweep) {
    for (int p = 0; p < 15; ++p) {
      for (int q = p + 1; q < 16; ++q) {
        if (l == 0) {
          double app = G[p][p], aqq = G[q][q], apq = G[p][q];
          double c = 1.0, s = 0.0;
          if (fabs(apq) > 1e-300) {
            double tau = (aqq - app) / (2.0 * apq);
            double t = ((tau >= 0.0) ? 1.0 : -1.0) / (fabs(tau) + sqrt(1.0 + tau * tau));
            c = 1.0 / sqrt(1.0 + t * t);
            s = t * c;
          }
          cs[0] = c; cs[1] = s;
        }
        __syncthreads();
        double c = cs[0], s = cs[1];
        if (l < 16) {
          double gjp = G[l][p], gjq = G[l][q];
          G[l][p] = c * gjp - s * gjq;
          G[l][q] = s * gjp + c * gjq;
        }
        __syncthreads();
        if (l < 16) {
          double gpj = G[p][l], gqj = G[q][l];
          G[p][l] = c * gpj - s * gqj;
          G[q][l] = s * gpj + c * gqj;
          double vjp = Vv[l][p], vjq = Vv[l][q];
          Vv[l][p] = c * vjp - s * vjq;
          Vv[l][q] = s * vjp + c * vjq;
        }
        __syncthreads();
      }
    }
  }
  if (l < 16) evals[l] = G[l][l];
  __syncthreads();
  if (l == 0) {
    int idx[16];
    for (int i = 0; i < 16; ++i) idx[i] = i;
    for (int i = 0; i < 16; ++i) {
      int best = i;
      for (int j = i + 1; j < 16; ++j)
        if (evals[idx[j]] > evals[idx[best]]) best = j;
      int t = idx[i]; idx[i] = idx[best]; idx[best] = t;
    }
    for (int i = 0; i < 16; ++i) perm[i] = idx[i];
  }
  __syncthreads();
  if (l < 16) {
    double w = evals[perm[l]];
    if (w < 1e-24) w = 1e-24;
    Sout[(size_t)mat * 16 + l] = (float)sqrt(w);
  }
  for (int i = l; i < 256; i += 32) {
    int d = i >> 4, e = i & 15;
    Vout[(size_t)mat * 256 + i] = (float)Vv[d][perm[e]];
  }
}

// ================= spectral cross-attention, one block per batch =================
__global__ void cross_attn_k(const float* __restrict__ Sin, float* __restrict__ Sout,
                             const float* __restrict__ lng, const float* __restrict__ lnb,
                             const float* __restrict__ qkvw, const float* __restrict__ qkvb,
                             const float* __restrict__ outw, const float* __restrict__ outb,
                             const float* __restrict__ alpha) {
  __shared__ float k_s[256][16];
  __shared__ float v_s[256][16];
  __shared__ float wq[48 * 16];
  __shared__ float bq[48];
  __shared__ float wo[16 * 16];
  __shared__ float bo[16], al[16], lg[16], lb[16];
  int n = threadIdx.x;
  for (int i = n; i < 48 * 16; i += 256) wq[i] = qkvw[i];
  if (n < 48) bq[n] = qkvb[n];
  if (n < 256) wo[n] = outw[n];
  if (n < 16) { bo[n] = outb[n]; al[n] = alpha[n]; lg[n] = lng[n]; lb[n] = lnb[n]; }
  __syncthreads();
  const float* srow = Sin + ((size_t)blockIdx.x * 256 + n) * 16;
  float s[16], u[16];
  float mean = 0.f;
  for (int d = 0; d < 16; ++d) { s[d] = srow[d]; mean += s[d]; }
  mean *= (1.0f / 16.0f);
  float var = 0.f;
  for (int d = 0; d < 16; ++d) { float t = s[d] - mean; var += t * t; }
  var *= (1.0f / 16.0f);
  float inv = rsqrtf(var + 1e-5f);
  for (int d = 0; d < 16; ++d) u[d] = (s[d] - mean) * inv * lg[d] + lb[d];
  float q[16];
  for (int o = 0; o < 48; ++o) {
    float acc = bq[o];
    for (int d = 0; d < 16; ++d) acc += wq[o * 16 + d] * u[d];
    if (o < 16) q[o] = acc;
    else if (o < 32) k_s[n][o - 16] = acc;
    else v_s[n][o - 32] = acc;
  }
  __syncthreads();
  float ou[16];
  for (int h = 0; h < 4; ++h) {
    float mx = -1e30f;
    for (int m = 0; m < 256; ++m) {
      float sc = 0.f;
      for (int d = 0; d < 4; ++d) sc += q[h * 4 + d] * k_s[m][h * 4 + d];
      sc *= 0.5f;
      mx = fmaxf(mx, sc);
    }
    float den = 0.f, oa0 = 0.f, oa1 = 0.f, oa2 = 0.f, oa3 = 0.f;
    for (int m = 0; m < 256; ++m) {
      float sc = 0.f;
      for (int d = 0; d < 4; ++d) sc += q[h * 4 + d] * k_s[m][h * 4 + d];
      float e = expf(sc * 0.5f - mx);
      den += e;
      oa0 += e * v_s[m][h * 4 + 0];
      oa1 += e * v_s[m][h * 4 + 1];
      oa2 += e * v_s[m][h * 4 + 2];
      oa3 += e * v_s[m][h * 4 + 3];
    }
    float r = 1.0f / den;
    ou[h * 4 + 0] = oa0 * r; ou[h * 4 + 1] = oa1 * r;
    ou[h * 4 + 2] = oa2 * r; ou[h * 4 + 3] = oa3 * r;
  }
  float* dst = Sout + ((size_t)blockIdx.x * 256 + n) * 16;
  for (int o = 0; o < 16; ++o) {
    float acc = bo[o];
    for (int d = 0; d < 16; ++d) acc += wo[o * 16 + d] * ou[d];
    float gate = tanhf(acc);
    float a = 0.2f / (1.0f + expf(-al[o]));
    dst[o] = s[o] * (1.0f + a * gate);
  }
}

// ================= mixing matrix: M = V diag(Snew/clip(S0)) V^T =================
__global__ void build_mix_k(const float* __restrict__ V, const float* __restrict__ S0,
                            const float* __restrict__ Snew, float* __restrict__ Mout) {
  int mat = blockIdx.x, t = threadIdx.x;
  __shared__ float Vr[256];
  __shared__ float r[16];
  Vr[t] = V[(size_t)mat * 256 + t];
  if (t < 16) {
    float s0 = S0[(size_t)mat * 16 + t];
    if (s0 < 1e-16f) s0 = 1e-16f;
    r[t] = Snew[(size_t)mat * 16 + t] / s0;
  }
  __syncthreads();
  int d = t >> 4, e2 = t & 15;
  float acc = 0.f;
  for (int e = 0; e < 16; ++e) acc += Vr[d * 16 + e] * r[e] * Vr[e2 * 16 + e];
  Mout[(size_t)mat * 256 + t] = acc;
}

// ================= A_rec = A(256x16) @ M(16x16) per matrix, f32 WMMA -> bf16 =================
__global__ __launch_bounds__(256) void arec_k(const float* __restrict__ A, const float* __restrict__ Mm,
                                              unsigned short* __restrict__ outb) {
  int wave = (blockIdx.x * blockDim.x + threadIdx.x) >> 5;
  int lane = threadIdx.x & 31;
  int mat = wave >> 4, rt = wave & 15;
  const float* Ab = A + (size_t)mat * 4096 + rt * 256;  // 16 rows x 16 cols
  const float* Mb = Mm + (size_t)mat * 256;
  int nloc = lane & 15, half = lane >> 4;
#if HAVE_WMMA_F32
  v8f acc = {};
#pragma unroll
  for (int k0 = 0; k0 < 16; k0 += 4) {
    v2f a = *(const v2f*)(Ab + nloc * 16 + k0 + 2 * half);  // b64 load
    v2f b;
    b.x = Mb[(k0 + 2 * half) * 16 + nloc];
    b.y = Mb[(k0 + 2 * half + 1) * 16 + nloc];
    acc = __builtin_amdgcn_wmma_f32_16x16x4_f32(false, a, false, b, (short)0, acc, false, false);
  }
#pragma unroll
  for (int i = 0; i < 8; ++i) {
    int m = rt * 16 + i + 8 * half;
    outb[(size_t)mat * 4096 + m * 16 + nloc] = f2bf(acc[i]);
  }
#else
  for (int i = lane; i < 256; i += 32) {
    int mr = i >> 4, nc = i & 15;
    float s = 0.f;
    for (int k = 0; k < 16; ++k) s += Ab[mr * 16 + k] * Mb[k * 16 + nc];
    outb[(size_t)mat * 4096 + (rt * 16 + mr) * 16 + nc] = f2bf(s);
  }
#endif
}

// ================= 3x3 SAME convs =================
template <int IC, int OC, bool GELU, bool RES>
__global__ void conv3x3_k(const float* __restrict__ in, const float* __restrict__ w,
                          const float* __restrict__ bias, const float* __restrict__ res,
                          float* __restrict__ out) {
  int idx = blockIdx.x * blockDim.x + threadIdx.x;
  if (idx >= 16 * OC * 65536) return;
  int x = idx & 255, y = (idx >> 8) & 255;
  int oc = (idx >> 16) % OC, b = idx / (OC * 65536);
  float acc = bias[oc];
#pragma unroll
  for (int ic = 0; ic < IC; ++ic) {
    const float* ip = in + ((size_t)(b * IC + ic)) * 65536;
    const float* wp = w + (oc * IC + ic) * 9;
#pragma unroll
    for (int ky = 0; ky < 3; ++ky) {
      int yy = y + ky - 1;
      if (yy < 0 || yy > 255) continue;
#pragma unroll
      for (int kx = 0; kx < 3; ++kx) {
        int xx = x + kx - 1;
        if (xx < 0 || xx > 255) continue;
        acc += ip[yy * 256 + xx] * wp[ky * 3 + kx];
      }
    }
  }
  if (GELU) acc = gelu_f(acc);
  if (RES) acc += res[idx];
  out[idx] = acc;
}

// ================= host orchestration =================
enum {
  IN_X = 0,
  IN_ENC_IN_W = 1, IN_ENC_IN_B = 2,
  IN_ENC_BLK = 3,                 // 4 blocks x {ln_g, ln_b, l1w, l1b, l2w, l2b}
  IN_ENC_OUT_W = 27, IN_ENC_OUT_B = 28,
  IN_DEC_IN_W = 29, IN_DEC_IN_B = 30,
  IN_DEC_BLK = 31,
  IN_DEC_OUT_W = 55, IN_DEC_OUT_B = 56,
  IN_CROSS = 57,                  // 2 x {ln_g, ln_b, qkvw, qkvb, outw, outb, alpha}
  IN_BS1_W = 71, IN_BS1_B = 72, IN_BS2_W = 73, IN_BS2_B = 74
};

static void run_gemm(const unsigned short* X, const unsigned short* W, const float* bias,
                     const float* res, float* of, unsigned short* ob, int M, int N, int K,
                     int mode, hipStream_t stream) {
  int blocks = (M / 128) * (N / 64);
  if (mode == 0)
    gemm_bf16_k<false, false, true, false><<<blocks, 256, 0, stream>>>(X, W, bias, res, of, ob, M, N, K);
  else if (mode == 1)
    gemm_bf16_k<true, false, false, true><<<blocks, 256, 0, stream>>>(X, W, bias, res, of, ob, M, N, K);
  else
    gemm_bf16_k<false, true, true, false><<<blocks, 256, 0, stream>>>(X, W, bias, res, of, ob, M, N, K);
}

extern "C" void kernel_launch(void* const* d_in, const int* in_sizes, int n_in,
                              void* d_out, int out_size, void* d_ws, size_t ws_size,
                              hipStream_t stream) {
  (void)in_sizes; (void)n_in; (void)out_size; (void)ws_size;
  char* ws = (char*)d_ws;
  size_t off = 0;
  auto carve = [&](size_t bytes) {
    void* p = (void*)(ws + off);
    off += (bytes + 255) & ~(size_t)255;
    return p;
  };
  const size_t M = 4096;  // B*N rows
  unsigned short* pt_bf   = (unsigned short*)carve(M * 768 * 2);
  float*          h_f     = (float*)carve(M * 768 * 4);
  unsigned short* u_bf    = (unsigned short*)carve(M * 768 * 2);
  unsigned short* t_bf    = (unsigned short*)carve(M * 768 * 2);
  float*          A_f     = (float*)carve(M * 4096 * 4);
  float*          G_f     = (float*)carve(M * 256 * 4);
  float*          V_f     = (float*)carve(M * 256 * 4);
  float*          S0_f    = (float*)carve(M * 16 * 4);
  float*          Sa_f    = (float*)carve(M * 16 * 4);
  float*          Sb_f    = (float*)carve(M * 16 * 4);
  float*          Mm_f    = (float*)carve(M * 256 * 4);
  unsigned short* Arec_bf = (unsigned short*)carve(M * 4096 * 2);
  float*          pat_f   = (float*)carve(M * 768 * 4);
  float*          img_f   = (float*)carve((size_t)16 * 3 * 65536 * 4);
  float*          y_f     = (float*)carve((size_t)16 * 16 * 65536 * 4);
  unsigned short* w_enc_in  = (unsigned short*)carve(768 * 768 * 2);
  unsigned short* w_enc_l1[4], *w_enc_l2[4], *w_dec_l1[4], *w_dec_l2[4];
  for (int i = 0; i < 4; ++i) {
    w_enc_l1[i] = (unsigned short*)carve(768 * 768 * 2);
    w_enc_l2[i] = (unsigned short*)carve(768 * 768 * 2);
  }
  unsigned short* w_enc_out = (unsigned short*)carve((size_t)4096 * 768 * 2);
  unsigned short* w_dec_in  = (unsigned short*)carve((size_t)768 * 4096 * 2);
  for (int i = 0; i < 4; ++i) {
    w_dec_l1[i] = (unsigned short*)carve(768 * 768 * 2);
    w_dec_l2[i] = (unsigned short*)carve(768 * 768 * 2);
  }
  unsigned short* w_dec_out = (unsigned short*)carve(768 * 768 * 2);

  auto cvt_in = [&](int idx, unsigned short* dst, int n) {
    cvt_f32_bf16_k<<<(n + 255) / 256, 256, 0, stream>>>((const float*)d_in[idx], dst, n);
  };
  // weight conversion (f32 -> bf16)
  cvt_in(IN_ENC_IN_W, w_enc_in, 768 * 768);
  for (int i = 0; i < 4; ++i) {
    cvt_in(IN_ENC_BLK + i * 6 + 2, w_enc_l1[i], 768 * 768);
    cvt_in(IN_ENC_BLK + i * 6 + 4, w_enc_l2[i], 768 * 768);
    cvt_in(IN_DEC_BLK + i * 6 + 2, w_dec_l1[i], 768 * 768);
    cvt_in(IN_DEC_BLK + i * 6 + 4, w_dec_l2[i], 768 * 768);
  }
  cvt_in(IN_ENC_OUT_W, w_enc_out, 4096 * 768);
  cvt_in(IN_DEC_IN_W, w_dec_in, 768 * 4096);
  cvt_in(IN_DEC_OUT_W, w_dec_out, 768 * 768);

  // patchify
  patchify_k<<<(4096 * 768 + 255) / 256, 256, 0, stream>>>((const float*)d_in[IN_X], pt_bf);

  // encoder
  run_gemm(pt_bf, w_enc_in, (const float*)d_in[IN_ENC_IN_B], nullptr, h_f, nullptr,
           4096, 768, 768, 0, stream);
  for (int i = 0; i < 4; ++i) {
    int base = IN_ENC_BLK + i * 6;
    layernorm_k<<<4096, 256, 0, stream>>>(h_f, (const float*)d_in[base], (const float*)d_in[base + 1], u_bf, 768);
    run_gemm(u_bf, w_enc_l1[i], (const float*)d_in[base + 3], nullptr, nullptr, t_bf, 4096, 768, 768, 1, stream);
    run_gemm(t_bf, w_enc_l2[i], (const float*)d_in[base + 5], h_f, h_f, nullptr, 4096, 768, 768, 2, stream);
  }
  cvt_f32_bf16_k<<<(4096 * 768 + 255) / 256, 256, 0, stream>>>(h_f, u_bf, 4096 * 768);
  run_gemm(u_bf, w_enc_out, (const float*)d_in[IN_ENC_OUT_B], nullptr, A_f, nullptr,
           4096, 4096, 768, 0, stream);

  // spectral path
  gram_k<<<4096 / 8, 256, 0, stream>>>(A_f, G_f);
  jacobi_eigh_k<<<4096, 32, 0, stream>>>(G_f, V_f, S0_f);
  {
    int b0 = IN_CROSS;
    cross_attn_k<<<16, 256, 0, stream>>>(S0_f, Sa_f,
        (const float*)d_in[b0], (const float*)d_in[b0 + 1], (const float*)d_in[b0 + 2],
        (const float*)d_in[b0 + 3], (const float*)d_in[b0 + 4], (const float*)d_in[b0 + 5],
        (const float*)d_in[b0 + 6]);
    int b1 = IN_CROSS + 7;
    cross_attn_k<<<16, 256, 0, stream>>>(Sa_f, Sb_f,
        (const float*)d_in[b1], (const float*)d_in[b1 + 1], (const float*)d_in[b1 + 2],
        (const float*)d_in[b1 + 3], (const float*)d_in[b1 + 4], (const float*)d_in[b1 + 5],
        (const float*)d_in[b1 + 6]);
  }
  build_mix_k<<<4096, 256, 0, stream>>>(V_f, S0_f, Sb_f, Mm_f);
  arec_k<<<(4096 * 16) / 8, 256, 0, stream>>>(A_f, Mm_f, Arec_bf);

  // decoder
  run_gemm(Arec_bf, w_dec_in, (const float*)d_in[IN_DEC_IN_B], nullptr, h_f, nullptr,
           4096, 768, 4096, 0, stream);
  for (int i = 0; i < 4; ++i) {
    int base = IN_DEC_BLK + i * 6;
    layernorm_k<<<4096, 256, 0, stream>>>(h_f, (const float*)d_in[base], (const float*)d_in[base + 1], u_bf, 768);
    run_gemm(u_bf, w_dec_l1[i], (const float*)d_in[base + 3], nullptr, nullptr, t_bf, 4096, 768, 768, 1, stream);
    run_gemm(t_bf, w_dec_l2[i], (const float*)d_in[base + 5], h_f, h_f, nullptr, 4096, 768, 768, 2, stream);
  }
  cvt_f32_bf16_k<<<(4096 * 768 + 255) / 256, 256, 0, stream>>>(h_f, u_bf, 4096 * 768);
  run_gemm(u_bf, w_dec_out, (const float*)d_in[IN_DEC_OUT_B], nullptr, pat_f, nullptr,
           4096, 768, 768, 0, stream);

  // stitch + boundary smoothing
  stitch_k<<<(16 * 3 * 65536 + 255) / 256, 256, 0, stream>>>(pat_f, img_f);
  conv3x3_k<3, 16, true, false><<<(16 * 16 * 65536 + 255) / 256, 256, 0, stream>>>(
      img_f, (const float*)d_in[IN_BS1_W], (const float*)d_in[IN_BS1_B], nullptr, y_f);
  conv3x3_k<16, 3, false, true><<<(16 * 3 * 65536 + 255) / 256, 256, 0, stream>>>(
      y_f, (const float*)d_in[IN_BS2_W], (const float*)d_in[IN_BS2_B], img_f, (float*)d_out);
}
